// TransformerBlock_22093311771187
// MI455X (gfx1250) — compile-verified, matrix operand order checked by reading, not run on hardware
//
#include <hip/hip_runtime.h>
#include <hip/hip_bf16.h>
#include <math.h>

typedef __attribute__((ext_vector_type(16))) __bf16 v16bf;
typedef __attribute__((ext_vector_type(8)))  __bf16 v8bf;
typedef __attribute__((ext_vector_type(8)))  float  v8f;

#define HID    1024
#define SEQ    2048
#define BATCH  2
#define NHEADS 8
#define HDIM   128
#define FF     4096
#define MROWS  (BATCH * SEQ)

#define BLK_M 128
#define BLK_N 128
#define BLK_K 64

// ---------------------------------------------------------------------------
// Fragment loader: CDNA5 16-bit A/B layout for 16x16x32 WMMA.
// Lane L (0..15): row = L, holds K = kb..kb+7 and kb+16..kb+23
// Lane L (16..31): row = L-16, holds K = kb+8..15 and kb+24..31
// => per lane: two contiguous 8-element (16B) chunks at +koff and +koff+16.
// ---------------------------------------------------------------------------
static __device__ __forceinline__ v16bf load_frag(const __bf16* p) {
  v8bf lo = *(const v8bf*)p;
  v8bf hi = *(const v8bf*)(p + 16);
  return __builtin_shufflevector(lo, hi, 0,1,2,3,4,5,6,7,8,9,10,11,12,13,14,15);
}

static __device__ __forceinline__ v8f wmma_bf16(v16bf a, v16bf b, v8f c) {
  return __builtin_amdgcn_wmma_f32_16x16x32_bf16(false, a, false, b, (short)0, c,
                                                 false, false);
}

// Async global -> LDS copy, 16B per lane, tracked by ASYNCcnt (gfx1250).
// lds_addr = low 32 bits of the flat shared-memory pointer.
static __device__ __forceinline__ void async_copy_b128(void* lds_ptr,
                                                       const __bf16* gptr) {
  uint32_t lds_addr = (uint32_t)(uintptr_t)lds_ptr;
  asm volatile("global_load_async_to_lds_b128 %0, %1, off"
               :: "v"(lds_addr), "v"((uint64_t)(uintptr_t)gptr)
               : "memory");
}

static __device__ __forceinline__ void wait_async0() {
  asm volatile("s_wait_asynccnt 0x0" ::: "memory");
}

// ---------------------------------------------------------------------------
// fp32 -> bf16 elementwise convert (weights)
// ---------------------------------------------------------------------------
__global__ void f32_to_bf16_kernel(const float* __restrict__ src,
                                   __bf16* __restrict__ dst, int n) {
  int i = blockIdx.x * blockDim.x + threadIdx.x;
  int stride = gridDim.x * blockDim.x;
  for (; i < n; i += stride) dst[i] = (__bf16)src[i];
}

// ---------------------------------------------------------------------------
// Row LayerNorm (H=1024), fp32 in -> bf16 out. One block (256 thr) per row.
// ---------------------------------------------------------------------------
__global__ __launch_bounds__(256)
void ln_bf16_kernel(const float* __restrict__ x, const float* __restrict__ g,
                    const float* __restrict__ b, __bf16* __restrict__ out) {
  const int row = blockIdx.x;
  const float* xr = x + (size_t)row * HID;
  float s = 0.f, s2 = 0.f;
  for (int i = threadIdx.x; i < HID; i += 256) {
    float v = xr[i];
    s += v; s2 += v * v;
  }
  __shared__ float ssum, ssq;
  if (threadIdx.x == 0) { ssum = 0.f; ssq = 0.f; }
  __syncthreads();
  #pragma unroll
  for (int d = 1; d < 32; d <<= 1) {
    s  += __shfl_xor(s,  d, 32);
    s2 += __shfl_xor(s2, d, 32);
  }
  if ((threadIdx.x & 31) == 0) { atomicAdd(&ssum, s); atomicAdd(&ssq, s2); }
  __syncthreads();
  const float mu  = ssum * (1.0f / HID);
  const float var = ssq * (1.0f / HID) - mu * mu;
  const float r   = rsqrtf(var + 1e-5f);
  for (int i = threadIdx.x; i < HID; i += 256)
    out[(size_t)row * HID + i] = (__bf16)((xr[i] - mu) * r * g[i] + b[i]);
}

// ---------------------------------------------------------------------------
// WMMA GEMM with async-LDS double-buffered staging.
// C[M,N] = A[M,K] (row-major) * W[N,K]^T (+ bias, epilogue modes).
// Block: 256 threads = 8 waves (4 M x 2 N); block tile 128x128;
// wave tile 32x64 (2x4 WMMA frags). K-step 64 (2 x 8 WMMAs per barrier pair).
// Staging: 128x64 bf16 tiles of A and W copied global->LDS with
// global_load_async_to_lds_b128 (ASYNCcnt), double buffered (64 KB LDS).
// mode 1: bf16 out in [B,heads,S,d]        (Q / K)
// mode 2: bf16 out in [B,heads,d,S]        (V^T)
// mode 3: fp32 out = resid + acc + bias    (O-proj / MLP2 residual)
// mode 4: bf16 out = gelu(acc + bias)      (MLP1)
// ---------------------------------------------------------------------------
__global__ __launch_bounds__(256)
void gemm_wmma_kernel(const __bf16* __restrict__ A, const __bf16* __restrict__ W,
                      const float* __restrict__ bias, const float* __restrict__ resid,
                      void* __restrict__ outp, int M, int N, int K, int mode) {
  (void)M;
  const int lane = threadIdx.x & 31;
  const int wave = threadIdx.x >> 5;
  const int wm   = wave & 3;          // 4 waves along M
  const int wn   = wave >> 2;         // 2 waves along N
  const int mblk = blockIdx.y * BLK_M;
  const int nblk = blockIdx.x * BLK_N;
  const int lrow = lane & 15;
  const int koff = (lane >> 4) << 3;  // 0 or 8

  __shared__ __align__(16) __bf16 tA[2][BLK_M * BLK_K];
  __shared__ __align__(16) __bf16 tB[2][BLK_N * BLK_K];

  // Stage one 128x64 K-slice of A and W into LDS buffer `buf`.
  // 1024 16B chunks per tile; 256 threads issue 4 chunks each per tile.
  auto stage = [&](int buf, int k0) {
    #pragma unroll
    for (int j = 0; j < 4; ++j) {
      const int id  = threadIdx.x + j * 256;
      const int row = id >> 3;
      const int ch  = (id & 7) << 3;  // element offset of the 16B chunk
      async_copy_b128(&tA[buf][row * BLK_K + ch],
                      A + (size_t)(mblk + row) * K + k0 + ch);
      async_copy_b128(&tB[buf][row * BLK_K + ch],
                      W + (size_t)(nblk + row) * K + k0 + ch);
    }
  };

  v8f acc[2][4] = {};

  stage(0, 0);
  for (int k0 = 0; k0 < K; k0 += BLK_K) {
    const int cur = (k0 >> 6) & 1;
    wait_async0();       // this wave's copies into tX[cur] have landed
    __syncthreads();     // everyone's copies have landed
    if (k0 + BLK_K < K) stage(cur ^ 1, k0 + BLK_K);

    #pragma unroll
    for (int ks = 0; ks < BLK_K; ks += 32) {
      v16bf a[2], b[4];
      #pragma unroll
      for (int mt = 0; mt < 2; ++mt)
        a[mt] = load_frag(&tA[cur][(wm * 32 + mt * 16 + lrow) * BLK_K + ks + koff]);
      #pragma unroll
      for (int nt = 0; nt < 4; ++nt)
        b[nt] = load_frag(&tB[cur][(wn * 64 + nt * 16 + lrow) * BLK_K + ks + koff]);
      #pragma unroll
      for (int mt = 0; mt < 2; ++mt)
        #pragma unroll
        for (int nt = 0; nt < 4; ++nt)
          acc[mt][nt] = wmma_bf16(a[mt], b[nt], acc[mt][nt]);
    }

    __syncthreads();     // all reads of tX[cur] done before it is re-staged
  }

  const int rplus = (lane >> 4) << 3;  // upper 16 lanes hold rows +8
  #pragma unroll
  for (int mt = 0; mt < 2; ++mt)
  #pragma unroll
  for (int nt = 0; nt < 4; ++nt)
  #pragma unroll
  for (int v = 0; v < 8; ++v) {
    const int row = mblk + wm * 32 + mt * 16 + v + rplus;
    const int col = nblk + wn * 64 + nt * 16 + lrow;
    float val = acc[mt][nt][v] + bias[col];
    if (mode == 1 || mode == 2) {
      const int bb = row >> 11, ss = row & (SEQ - 1);   // S = 2048
      const int hh = col >> 7,  dd = col & (HDIM - 1);  // d = 128
      __bf16* o = (__bf16*)outp;
      size_t idx = (mode == 1)
          ? ((size_t)((bb * NHEADS + hh) * SEQ + ss)) * HDIM + dd
          : ((size_t)((bb * NHEADS + hh) * HDIM + dd)) * SEQ + ss;
      o[idx] = (__bf16)val;
    } else if (mode == 3) {
      float* o = (float*)outp;
      o[(size_t)row * N + col] = resid[(size_t)row * N + col] + val;
    } else {  // mode 4: exact GELU, bf16 out
      __bf16* o = (__bf16*)outp;
      float gl = 0.5f * val * (1.0f + erff(val * 0.70710678118654752f));
      o[(size_t)row * N + col] = (__bf16)gl;
    }
  }
}

// ---------------------------------------------------------------------------
// Causal flash attention with shared async-LDS K/V staging.
// Block: 128 threads = 4 waves; each wave owns a 16-query tile (64 rows/block).
// Per 32-key block: K tile (32x128) and V^T tile (128x32) staged global->LDS
// (double buffered, ASYNCcnt) once and consumed by all 4 waves.
// All waves run to the block's last key block; the causal mask nulls any
// key > query row (exp(-huge)=0), so lower waves' surplus blocks are inert.
// ---------------------------------------------------------------------------
__global__ __launch_bounds__(128)
void attn_wmma_kernel(const __bf16* __restrict__ Q, const __bf16* __restrict__ Kc,
                      const __bf16* __restrict__ VT, __bf16* __restrict__ Y) {
  const int lane  = threadIdx.x & 31;
  const int wave  = threadIdx.x >> 5;
  const int lrow  = lane & 15;
  const int koff  = (lane >> 4) << 3;
  const int rplus = koff;               // C-layout: upper lanes own rows +8
  const int qblk  = blockIdx.x * 64;
  const int qbase = qblk + wave * 16;
  const int bh    = blockIdx.y;
  const float scale = 0.08838834764831845f;  // 1/sqrt(128)
  const float NEGF  = -3.0e38f;

  __shared__ __align__(16) __bf16 kt[2][32 * HDIM];   // 2 x 8 KB
  __shared__ __align__(16) __bf16 vt[2][HDIM * 32];   // 2 x 8 KB
  __shared__ __align__(16) __bf16 pbuf[4][16 * 32];   // per-wave P tiles

  // Stage K tile (32 keys x 128) and V^T tile (128 x 32 keys) for key block kb.
  // 512 16B chunks per tile; 128 threads issue 4 chunks each per tile.
  auto stage = [&](int buf, int kb) {
    #pragma unroll
    for (int j = 0; j < 4; ++j) {
      const int id = threadIdx.x + j * 128;
      const int kr = id >> 4, kc = (id & 15) << 3;   // K tile: 16 chunks/row
      async_copy_b128(&kt[buf][kr * HDIM + kc],
                      Kc + ((size_t)bh * SEQ + kb + kr) * HDIM + kc);
      const int vr = id >> 2, vc = (id & 3) << 3;    // V^T tile: 4 chunks/row
      async_copy_b128(&vt[buf][vr * 32 + vc],
                      VT + ((size_t)bh * HDIM + vr) * SEQ + kb + vc);
    }
  };

  v16bf qf[4];
  #pragma unroll
  for (int kc = 0; kc < 4; ++kc)
    qf[kc] = load_frag(Q + ((size_t)bh * SEQ + qbase + lrow) * HDIM + kc * 32 + koff);

  v8f o[8] = {};
  float mrow[8], lsum[8];
  #pragma unroll
  for (int v = 0; v < 8; ++v) { mrow[v] = NEGF; lsum[v] = 0.f; }

  const int kend = qblk + 64;           // block-uniform key range
  stage(0, 0);
  for (int kb = 0; kb < kend; kb += 32) {
    const int cur = (kb >> 5) & 1;
    wait_async0();
    __syncthreads();
    if (kb + 32 < kend) stage(cur ^ 1, kb + 32);

    // ---- S = Q K^T over d=128 (4 x K=32 WMMAs per 16-key column tile) ----
    v8f s0acc = {}, s1acc = {};
    #pragma unroll
    for (int kc = 0; kc < 4; ++kc) {
      v16bf k0 = load_frag(&kt[cur][(     lrow) * HDIM + kc * 32 + koff]);
      v16bf k1 = load_frag(&kt[cur][(16 + lrow) * HDIM + kc * 32 + koff]);
      s0acc = wmma_bf16(qf[kc], k0, s0acc);
      s1acc = wmma_bf16(qf[kc], k1, s1acc);
    }
    // ---- scale, causal mask, online softmax (per C-layout row) ----
    #pragma unroll
    for (int v = 0; v < 8; ++v) {
      const int qrow = qbase + v + rplus;
      float s0 = s0acc[v] * scale;
      float s1 = s1acc[v] * scale;
      if (kb +      lrow > qrow) s0 = NEGF;
      if (kb + 16 + lrow > qrow) s1 = NEGF;
      float t = fmaxf(s0, s1);
      #pragma unroll
      for (int d = 1; d < 16; d <<= 1) t = fmaxf(t, __shfl_xor(t, d, 32));
      const float mnew = fmaxf(mrow[v], t);
      const float sc   = __expf(mrow[v] - mnew);
      const float p0   = __expf(s0 - mnew);
      const float p1   = __expf(s1 - mnew);
      float rs = p0 + p1;
      #pragma unroll
      for (int d = 1; d < 16; d <<= 1) rs += __shfl_xor(rs, d, 32);
      lsum[v] = lsum[v] * sc + rs;
      mrow[v] = mnew;
      #pragma unroll
      for (int dt = 0; dt < 8; ++dt) o[dt][v] *= sc;
      const int prow = v + rplus;
      pbuf[wave][prow * 32 +      lrow] = (__bf16)p0;
      pbuf[wave][prow * 32 + 16 + lrow] = (__bf16)p1;
    }
    __syncthreads();
    v16bf pf = load_frag(&pbuf[wave][lrow * 32 + koff]);
    // ---- O += P V (keys as contraction dim; V^T rows are contiguous) ----
    #pragma unroll
    for (int dt = 0; dt < 8; ++dt) {
      v16bf vf = load_frag(&vt[cur][(dt * 16 + lrow) * 32 + koff]);
      o[dt] = wmma_bf16(pf, vf, o[dt]);
    }
    __syncthreads();     // kt/vt[cur] + pbuf free before re-staging / rewrite
  }
  // ---- normalize and scatter back to [B,S,HID] ----
  const int b = bh >> 3, h = bh & 7;
  #pragma unroll
  for (int dt = 0; dt < 8; ++dt)
  #pragma unroll
  for (int v = 0; v < 8; ++v) {
    const int row = qbase + v + rplus;
    const int col = h * HDIM + dt * 16 + lrow;
    Y[((size_t)b * SEQ + row) * HID + col] = (__bf16)(o[dt][v] / lsum[v]);
  }
}

// ---------------------------------------------------------------------------
// Orchestration
// ---------------------------------------------------------------------------
extern "C" void kernel_launch(void* const* d_in, const int* in_sizes, int n_in,
                              void* d_out, int out_size, void* d_ws, size_t ws_size,
                              hipStream_t stream) {
  (void)in_sizes; (void)n_in; (void)out_size; (void)ws_size;

  const float* x    = (const float*)d_in[0];
  const float* ln1g = (const float*)d_in[1];
  const float* ln1b = (const float*)d_in[2];
  const float* wq   = (const float*)d_in[3];
  const float* bq   = (const float*)d_in[4];
  const float* wk   = (const float*)d_in[5];
  const float* bk   = (const float*)d_in[6];
  const float* wv   = (const float*)d_in[7];
  const float* bv   = (const float*)d_in[8];
  const float* wo   = (const float*)d_in[9];
  const float* bo   = (const float*)d_in[10];
  const float* ln2g = (const float*)d_in[11];
  const float* ln2b = (const float*)d_in[12];
  const float* w1   = (const float*)d_in[13];
  const float* b1   = (const float*)d_in[14];
  const float* w2   = (const float*)d_in[15];
  const float* b2   = (const float*)d_in[16];
  float* out = (float*)d_out;

  char* ws = (char*)d_ws;
  size_t off = 0;
  auto take = [&](size_t bytes) -> char* {
    char* p = ws + off;
    off += bytes;
    off = (off + 255) & ~(size_t)255;
    return p;
  };

  __bf16* wqb = (__bf16*)take((size_t)HID * HID * 2);
  __bf16* wkb = (__bf16*)take((size_t)HID * HID * 2);
  __bf16* wvb = (__bf16*)take((size_t)HID * HID * 2);
  __bf16* wob = (__bf16*)take((size_t)HID * HID * 2);
  __bf16* w1b = (__bf16*)take((size_t)FF  * HID * 2);
  __bf16* w2b = (__bf16*)take((size_t)HID * FF  * 2);
  __bf16* hn  = (__bf16*)take((size_t)MROWS * HID * 2);  // LN1 out
  __bf16* qb  = (__bf16*)take((size_t)MROWS * HID * 2);  // [B,h,S,d]
  __bf16* kbf = (__bf16*)take((size_t)MROWS * HID * 2);  // [B,h,S,d]
  __bf16* vtb = (__bf16*)take((size_t)MROWS * HID * 2);  // [B,h,d,S]
  __bf16* yb  = (__bf16*)take((size_t)MROWS * HID * 2);  // attn out [B,S,H]
  float*  x1  = (float*) take((size_t)MROWS * HID * 4);  // x + attn
  __bf16* m2  = (__bf16*)take((size_t)MROWS * HID * 2);  // LN2 out
  __bf16* gb  = (__bf16*)take((size_t)MROWS * FF  * 2);  // GELU(MLP1)

  // weights -> bf16
  f32_to_bf16_kernel<<<1024, 256, 0, stream>>>(wq, wqb, HID * HID);
  f32_to_bf16_kernel<<<1024, 256, 0, stream>>>(wk, wkb, HID * HID);
  f32_to_bf16_kernel<<<1024, 256, 0, stream>>>(wv, wvb, HID * HID);
  f32_to_bf16_kernel<<<1024, 256, 0, stream>>>(wo, wob, HID * HID);
  f32_to_bf16_kernel<<<2048, 256, 0, stream>>>(w1, w1b, FF * HID);
  f32_to_bf16_kernel<<<2048, 256, 0, stream>>>(w2, w2b, HID * FF);

  // LN1
  ln_bf16_kernel<<<MROWS, 256, 0, stream>>>(x, ln1g, ln1b, hn);

  // QKV projections (block tile 128x128, K-step 64)
  dim3 gH(HID / BLK_N, MROWS / BLK_M);
  gemm_wmma_kernel<<<gH, 256, 0, stream>>>(hn, wqb, bq, nullptr, qb,  MROWS, HID, HID, 1);
  gemm_wmma_kernel<<<gH, 256, 0, stream>>>(hn, wkb, bk, nullptr, kbf, MROWS, HID, HID, 1);
  gemm_wmma_kernel<<<gH, 256, 0, stream>>>(hn, wvb, bv, nullptr, vtb, MROWS, HID, HID, 2);

  // causal flash attention (64 query rows per block, shared K/V staging)
  dim3 gA(SEQ / 64, BATCH * NHEADS);
  attn_wmma_kernel<<<gA, 128, 0, stream>>>(qb, kbf, vtb, yb);

  // O-proj + residual -> x1 (fp32)
  gemm_wmma_kernel<<<gH, 256, 0, stream>>>(yb, wob, bo, x, x1, MROWS, HID, HID, 3);

  // LN2
  ln_bf16_kernel<<<MROWS, 256, 0, stream>>>(x1, ln2g, ln2b, m2);

  // MLP
  dim3 gF(FF / BLK_N, MROWS / BLK_M);
  gemm_wmma_kernel<<<gF, 256, 0, stream>>>(m2, w1b, b1, nullptr, gb, MROWS, FF, HID, 4);
  gemm_wmma_kernel<<<gH, 256, 0, stream>>>(gb, w2b, b2, x1, out, MROWS, HID, FF, 3);
}
